// Latent_SSM_Irregular_Mesh_23845658428252
// MI455X (gfx1250) — matrix operand order repaired and checked by reading, not compiled, not run on hardware
//
#include <hip/hip_runtime.h>
#include <stdint.h>

// ---------------- problem constants ----------------
#define B_      2
#define N_MESH  100000
#define G_      64          // L
#define HEADS_  8
#define DH_     64
#define DIM_    256
#define D_MODEL 512
#define D_STATE 64
#define D_INNER 1024
#define NHEADS  16
#define HEADDIM 64
#define CONV_DIM 1152       // D_INNER + 2*D_STATE
#define D_IN_PROJ 2208      // 2*D_INNER + 2*D_STATE + 2*NHEADS

// ---------------- WMMA types ----------------
typedef __attribute__((ext_vector_type(16))) __bf16   bf16x16;
typedef __attribute__((ext_vector_type(8)))  float    floatx8;
typedef __attribute__((ext_vector_type(4)))  uint32_t uint32x4;

union Frag16 { bf16x16 v; uint32x4 q[2]; };

__device__ __forceinline__ uint32_t packbf(float a, float b) {
    uint32_t ua = __float_as_uint(a), ub = __float_as_uint(b);
    ua = (ua + 0x7FFFu + ((ua >> 16) & 1u)) >> 16;
    ub = (ub + 0x7FFFu + ((ub >> 16) & 1u)) >> 16;
    return (ua & 0xFFFFu) | (ub << 16);
}
__device__ __forceinline__ uint16_t f2bf16(float a) {
    uint32_t ua = __float_as_uint(a);
    return (uint16_t)((ua + 0x7FFFu + ((ua >> 16) & 1u)) >> 16);
}
__device__ __forceinline__ float siluf(float x) { return x * (1.0f / (1.0f + expf(-x))); }
__device__ __forceinline__ float softplusf(float x) { return (x > 20.0f) ? x : log1pf(expf(x)); }

// ---------------- CDNA5 async memory -> LDS (ASYNCcnt-tracked) ----------------
// Copies 16B from global memory straight into LDS, no VGPR round-trip.
__device__ __forceinline__ void async_copy16(const void* gaddr, void* lds_generic) {
    // low 32 bits of a generic pointer into LDS == LDS byte offset on amdgcn
    uint32_t loff = (uint32_t)(uintptr_t)lds_generic;
    asm volatile("global_load_async_to_lds_b128 %0, %1, off"
                 :: "v"(loff), "v"(gaddr) : "memory");
}
__device__ __forceinline__ void wait_async0() {
    asm volatile("s_wait_asynccnt 0" ::: "memory");
}

// ================= K1: in_proj GEMM  u(2,64,512) @ W^T(512,2208) =================
// u[b,g,k] = slice_token[b, k/64, g, k%64]
__global__ void k_in_proj(const float* __restrict__ st, const float* __restrict__ W,
                          float* __restrict__ zxbcdt) {
    __shared__ float u_s[D_MODEL];
    const int row = blockIdx.x;              // 0..127  (b*64+g)
    const int b = row >> 6, g = row & 63;
    for (int k = threadIdx.x; k < D_MODEL; k += 256)
        u_s[k] = st[(((size_t)b * HEADS_ + (k >> 6)) * G_ + g) * DH_ + (k & 63)];
    __syncthreads();
    const int col = blockIdx.y * 256 + threadIdx.x;
    if (col >= D_IN_PROJ) return;
    const float* wr = W + (size_t)col * D_MODEL;
    float acc = 0.f;
    #pragma unroll 4
    for (int k = 0; k < D_MODEL; k += 4) {
        float4 wv = *(const float4*)(wr + k);
        float4 uv = *(const float4*)(u_s + k);
        acc += wv.x * uv.x + wv.y * uv.y + wv.z * uv.z + wv.w * uv.w;
    }
    zxbcdt[(size_t)row * D_IN_PROJ + col] = acc;
}

// ================= K2: dt = softplus(concat(dt_fwd, flip(dt_bwd)) + bias) =========
__global__ void k_dt(const float* __restrict__ zxbcdt, const float* __restrict__ dt_bias,
                     float* __restrict__ dtb) {
    const int idx = blockIdx.x * 256 + threadIdx.x;       // 4*64*16
    if (idx >= 4 * G_ * NHEADS) return;
    const int h = idx & 15, t = (idx >> 4) & 63, bb = idx >> 10;
    const int b = bb & 1, rev = bb >> 1;
    const int tt = rev ? (G_ - 1 - t) : t;
    const int col = D_INNER + CONV_DIM + (rev ? NHEADS : 0) + h;   // 2176 + ...
    float v = zxbcdt[((size_t)b * G_ + tt) * D_IN_PROJ + col] + dt_bias[h];
    dtb[idx] = softplusf(v);
}

// ================= K3: depthwise conv(3, pad 1) + SiLU over xBC ===================
__global__ void k_conv(const float* __restrict__ zxbcdt, const float* __restrict__ cw,
                       const float* __restrict__ cb, float* __restrict__ xbc) {
    const int idx = blockIdx.x * 256 + threadIdx.x;       // 2*64*1152
    if (idx >= B_ * G_ * CONV_DIM) return;
    const int ch = idx % CONV_DIM;
    const int t = (idx / CONV_DIM) % G_;
    const int b = idx / (CONV_DIM * G_);
    float acc = cb[ch];
    #pragma unroll
    for (int j = 0; j < 3; ++j) {
        int tt = t - 1 + j;
        float v = (tt >= 0 && tt < G_)
                ? zxbcdt[((size_t)b * G_ + tt) * D_IN_PROJ + D_INNER + ch] : 0.f;
        acc = fmaf(v, cw[ch * 3 + j], acc);
    }
    xbc[((size_t)b * G_ + t) * CONV_DIM + ch] = siluf(acc);
}

// ================= K4: SSD scan (4,L,16,64) — thread per (bb,h,p) ================
__global__ void k_scan(const float* __restrict__ xbc, const float* __restrict__ dtb,
                       const float* __restrict__ A_log, float* __restrict__ yscan) {
    const int blk = blockIdx.x;               // 0..63 = bb*16 + h
    const int bb = blk >> 4, h = blk & 15;
    const int p = threadIdx.x;                // 0..63
    const int b = bb & 1, rev = bb >> 1;
    const float A = -expf(A_log[h]);
    float hs[D_STATE];
    #pragma unroll
    for (int n = 0; n < D_STATE; ++n) hs[n] = 0.f;
    for (int t = 0; t < G_; ++t) {
        const int tt = rev ? (G_ - 1 - t) : t;
        const float dtv = dtb[((size_t)bb * G_ + t) * NHEADS + h];
        const size_t rb = ((size_t)b * G_ + tt) * CONV_DIM;
        const float xv = xbc[rb + h * HEADDIM + p];
        const float dA = expf(dtv * A);
        const float dtx = dtv * xv;
        const float* Brow = xbc + rb + D_INNER;
        const float* Crow = Brow + D_STATE;
        float y = 0.f;
        #pragma unroll
        for (int n = 0; n < D_STATE; ++n) {
            hs[n] = fmaf(dA, hs[n], dtx * Brow[n]);
            y = fmaf(hs[n], Crow[n], y);
        }
        yscan[(((size_t)bb * G_ + t) * NHEADS + h) * HEADDIM + p] = y;
    }
}

// ================= K5: diag = x @ fc_D_w^T + D ====================================
__global__ void k_diag(const float* __restrict__ xbc, const float* __restrict__ fcD,
                       const float* __restrict__ Dv, float* __restrict__ diag) {
    const int idx = blockIdx.x * 256 + threadIdx.x;        // 2*64*16
    if (idx >= B_ * G_ * NHEADS) return;
    const int h = idx & 15, t = (idx >> 4) & 63, b = idx >> 10;
    const float* xr = xbc + ((size_t)b * G_ + t) * CONV_DIM;
    const float* wr = fcD + (size_t)h * D_INNER;
    float acc = Dv[h];
    #pragma unroll 4
    for (int d = 0; d < D_INNER; ++d) acc = fmaf(xr[d], wr[d], acc);
    diag[idx] = acc;
}

// ================= K6a: shift/flip combine + skip + gate + RMSNorm ===============
__global__ void k_combine(const float* __restrict__ yscan, const float* __restrict__ xbc,
                          const float* __restrict__ diag, const float* __restrict__ zxbcdt,
                          const float* __restrict__ norm_w, float* __restrict__ yn) {
    __shared__ float red[256];
    const int row = blockIdx.x;                // b*64+t
    const int b = row >> 6, t = row & 63;
    const int tid = threadIdx.x;
    float yv4[4];
    float ss = 0.f;
    #pragma unroll
    for (int k = 0; k < 4; ++k) {
        const int d = tid + 256 * k;
        const int h = d >> 6, p = d & 63;
        float yf = (t > 0)
            ? yscan[(((size_t)b * G_ + (t - 1)) * NHEADS + h) * HEADDIM + p] : 0.f;
        float yb = (t < G_ - 1)
            ? yscan[((((size_t)(b + 2)) * G_ + (G_ - 2 - t)) * NHEADS + h) * HEADDIM + p] : 0.f;
        float x = xbc[(size_t)row * CONV_DIM + d];
        float yv = yf + yb + x * diag[(size_t)row * NHEADS + h];
        float z = zxbcdt[(size_t)row * D_IN_PROJ + d];
        yv *= siluf(z);
        yv4[k] = yv;
        ss = fmaf(yv, yv, ss);
    }
    red[tid] = ss; __syncthreads();
    for (int off = 128; off > 0; off >>= 1) {
        if (tid < off) red[tid] += red[tid + off];
        __syncthreads();
    }
    const float r = rsqrtf(red[0] * (1.0f / D_INNER) + 1e-5f);
    #pragma unroll
    for (int k = 0; k < 4; ++k) {
        const int d = tid + 256 * k;
        yn[(size_t)row * D_INNER + d] = yv4[k] * r * norm_w[d];
    }
}

// ================= K6b: out = yn @ out_proj_w^T  (128 x 512, K=1024) =============
__global__ void k_outproj(const float* __restrict__ yn, const float* __restrict__ W,
                          float* __restrict__ ost) {
    const int idx = blockIdx.x * 256 + threadIdx.x;        // 2*64*512
    if (idx >= B_ * G_ * D_MODEL) return;
    const int m = idx & 511, row = idx >> 9;
    const float* yr = yn + (size_t)row * D_INNER;
    const float* wr = W + (size_t)m * D_INNER;
    float acc = 0.f;
    #pragma unroll 4
    for (int d = 0; d < D_INNER; d += 4) {
        float4 a = *(const float4*)(yr + d);
        float4 w = *(const float4*)(wr + d);
        acc += a.x * w.x + a.y * w.y + a.z * w.z + a.w * w.w;
    }
    ost[idx] = acc;
}

// ================= K7: M_t[b][d][k=h*64+g] = sum_c ost[b,g,h*64+c]*to_out_w[d,h*64+c]
__global__ void k_makeM(const float* __restrict__ ost, const float* __restrict__ tw,
                        uint16_t* __restrict__ Mbf) {
    const int idx = blockIdx.x * 256 + threadIdx.x;        // 2*256*512
    if (idx >= B_ * DIM_ * 512) return;
    const int k = idx & 511, d = (idx >> 9) & 255, b = idx >> 17;
    const int h = k >> 6, g = k & 63;
    const float* orow = ost + ((size_t)b * G_ + g) * D_MODEL + h * 64;
    const float* wrow = tw + (size_t)d * D_MODEL + h * 64;
    float acc = 0.f;
    #pragma unroll 4
    for (int c = 0; c < 64; ++c) acc = fmaf(orow[c], wrow[c], acc);
    Mbf[idx] = f2bf16(acc);
}

// ================= K8: big fused GEMM:  (100000 x 512) @ (512 x 256) per batch ====
// A = slice_weights[b,h,n,g] (k=h*64+g, fp32 -> bf16 at frag build), B = M_t (bf16).
// Async global->LDS double-buffered pipeline; per wave: 2 A-frags x 8 B-frags.
#define A_STRIDE 40   // floats per LDS A row (32 + pad), 160B: b128-aligned, 2-way max
#define B_STRIDE 24   // dwords per LDS B row (16 + pad), 96B: b128-aligned, 2-way max

__device__ __forceinline__ void build_afrag(const float* lAf, int r, int half, Frag16& af) {
    const float4* p0 = (const float4*)(lAf + r * A_STRIDE + half * 8);
    const float4* p1 = (const float4*)(lAf + r * A_STRIDE + 16 + half * 8);
    float4 a = p0[0], b = p0[1], c = p1[0], d = p1[1];
    uint32x4 q0 = { packbf(a.x, a.y), packbf(a.z, a.w), packbf(b.x, b.y), packbf(b.z, b.w) };
    uint32x4 q1 = { packbf(c.x, c.y), packbf(c.z, c.w), packbf(d.x, d.y), packbf(d.z, d.w) };
    af.q[0] = q0; af.q[1] = q1;
}

__global__ __launch_bounds__(256) void k_slice_gemm(
    const float* __restrict__ sw, const uint16_t* __restrict__ Mbf,
    const float* __restrict__ bias, float* __restrict__ out) {
    const int b = blockIdx.y;
    const long n0 = (long)blockIdx.x * 128;
    const int tid = threadIdx.x;
    const int wave = tid >> 5, lane = tid & 31;
    const int half = lane >> 4, nl = lane & 15;
    const int wrow = wave & 3;       // 4 row-groups of 32 rows
    const int wcol = wave >> 2;      // 2 col-groups of 128 cols

    __shared__ float    lAf[2][128 * A_STRIDE];   // fp32 A slices (raw async copy)
    __shared__ uint32_t lBp[2][256 * B_STRIDE];   // packed bf16-pair B slices

    floatx8 acc[2][8];
    #pragma unroll
    for (int rt = 0; rt < 2; ++rt)
        #pragma unroll
        for (int j = 0; j < 8; ++j) acc[rt][j] = (floatx8)(0.0f);

    const float* swb = sw + (size_t)b * HEADS_ * N_MESH * 64;
    const uint16_t* Mb = Mbf + (size_t)b * DIM_ * 512;

    auto issue_slice = [&](int s, int buf) {
        const int hh = s >> 1, g0 = (s & 1) * 32;
        // A: 128 rows x 32 k fp32 (16 KB) : 1024 x 16B async copies
        #pragma unroll
        for (int it = 0; it < 4; ++it) {
            const int item = tid + it * 256;
            const int m = item >> 3, q = item & 7;
            long row = n0 + m; if (row >= N_MESH) row = N_MESH - 1;   // clamp (stores masked)
            const float* g = swb + ((size_t)hh * N_MESH + row) * 64 + g0 + q * 4;
            async_copy16(g, &lAf[buf][m * A_STRIDE + q * 4]);
        }
        // B: 256 d-rows x 32 k bf16 (16 KB) : 1024 x 16B async copies
        #pragma unroll
        for (int it = 0; it < 4; ++it) {
            const int item = tid + it * 256;
            const int d = item >> 2, q = item & 3;
            const uint16_t* g = Mb + (size_t)d * 512 + s * 32 + q * 8;
            async_copy16(g, &lBp[buf][d * B_STRIDE + q * 4]);
        }
    };

    issue_slice(0, 0);

    for (int s = 0; s < 16; ++s) {               // K steps of 32
        const int buf = s & 1;
        wait_async0();                           // this wave's slice-s copies landed
        __syncthreads();                         // everyone's copies landed, prev compute done
        if (s + 1 < 16) issue_slice(s + 1, buf ^ 1);   // overlap next slice with compute

        Frag16 a0, a1;
        build_afrag(&lAf[buf][0], wrow * 32 + nl,      half, a0);
        build_afrag(&lAf[buf][0], wrow * 32 + 16 + nl, half, a1);
        #pragma unroll
        for (int j = 0; j < 8; ++j) {
            Frag16 bf;
            const uint32_t* bbp = &lBp[buf][(wcol * 128 + j * 16 + nl) * B_STRIDE + half * 4];
            bf.q[0] = *(const uint32x4*)(bbp);
            bf.q[1] = *(const uint32x4*)(bbp + 8);
            acc[0][j] = __builtin_amdgcn_wmma_f32_16x16x32_bf16(
                false, a0.v, false, bf.v, (short)0, acc[0][j], false, false);
            acc[1][j] = __builtin_amdgcn_wmma_f32_16x16x32_bf16(
                false, a1.v, false, bf.v, (short)0, acc[1][j], false, false);
        }
    }

    // store C: VGPR r -> M = half*8 + r ; lane%16 -> N
    #pragma unroll
    for (int rt = 0; rt < 2; ++rt) {
        #pragma unroll
        for (int j = 0; j < 8; ++j) {
            const int d = wcol * 128 + j * 16 + nl;
            const float bv = bias[d];
            #pragma unroll
            for (int r = 0; r < 8; ++r) {
                const long row = n0 + wrow * 32 + rt * 16 + half * 8 + r;
                if (row < N_MESH)
                    out[((size_t)b * N_MESH + row) * DIM_ + d] = acc[rt][j][r] + bv;
            }
        }
    }
}

// =================================================================================
extern "C" void kernel_launch(void* const* d_in, const int* in_sizes, int n_in,
                              void* d_out, int out_size, void* d_ws, size_t ws_size,
                              hipStream_t stream) {
    const float* slice_token   = (const float*)d_in[0];
    const float* slice_weights = (const float*)d_in[1];
    const float* in_proj_w     = (const float*)d_in[2];
    const float* conv_w        = (const float*)d_in[3];
    const float* conv_b        = (const float*)d_in[4];
    const float* dt_bias       = (const float*)d_in[5];
    const float* A_log         = (const float*)d_in[6];
    const float* fc_D_w        = (const float*)d_in[7];
    const float* Dv            = (const float*)d_in[8];
    const float* norm_w        = (const float*)d_in[9];
    const float* out_proj_w    = (const float*)d_in[10];
    const float* to_out_w      = (const float*)d_in[11];
    const float* to_out_b      = (const float*)d_in[12];
    float* out = (float*)d_out;

    // workspace layout (floats)
    float* ws      = (float*)d_ws;
    float* zxbcdt  = ws;                      // 2*64*2208 = 282624
    float* dtb     = zxbcdt + 282624;         // 4*64*16   = 4096
    float* xbc     = dtb + 4096;              // 2*64*1152 = 147456
    float* yscan   = xbc + 147456;            // 4*64*16*64= 262144
    float* diag    = yscan + 262144;          // 2*64*16   = 2048
    float* yn      = diag + 2048;             // 2*64*1024 = 131072
    float* ost     = yn + 131072;             // 2*64*512  = 65536
    uint16_t* Mbf  = (uint16_t*)(ost + 65536);// 2*256*512 bf16

    k_in_proj<<<dim3(B_ * G_, (D_IN_PROJ + 255) / 256), 256, 0, stream>>>(slice_token, in_proj_w, zxbcdt);
    k_dt<<<(4 * G_ * NHEADS + 255) / 256, 256, 0, stream>>>(zxbcdt, dt_bias, dtb);
    k_conv<<<(B_ * G_ * CONV_DIM + 255) / 256, 256, 0, stream>>>(zxbcdt, conv_w, conv_b, xbc);
    k_scan<<<64, 64, 0, stream>>>(xbc, dtb, A_log, yscan);
    k_diag<<<(B_ * G_ * NHEADS + 255) / 256, 256, 0, stream>>>(xbc, fc_D_w, Dv, diag);
    k_combine<<<B_ * G_, 256, 0, stream>>>(yscan, xbc, diag, zxbcdt, norm_w, yn);
    k_outproj<<<(B_ * G_ * D_MODEL + 255) / 256, 256, 0, stream>>>(yn, out_proj_w, ost);
    k_makeM<<<(B_ * DIM_ * 512 + 255) / 256, 256, 0, stream>>>(ost, to_out_w, Mbf);

    const int ntiles = (N_MESH + 127) / 128;  // 782
    k_slice_gemm<<<dim3(ntiles, B_), 256, 0, stream>>>(slice_weights, Mbf, to_out_b, out);
    (void)in_sizes; (void)n_in; (void)out_size; (void)ws_size;
}
